// EncoderBlock_12232066859065
// MI455X (gfx1250) — matrix operand order repaired and checked by reading, not compiled, not run on hardware
//
#include <hip/hip_runtime.h>
#include <math.h>

typedef __bf16 bf16_t;
typedef __attribute__((ext_vector_type(16))) __bf16 v16bf;
typedef __attribute__((ext_vector_type(8)))  __bf16 v8bf;
typedef __attribute__((ext_vector_type(8)))  float  v8f;

#define DEV static __device__ __forceinline__

DEV float gelu_f(float x) {
    const float c = 0.7978845608028654f;   // sqrt(2/pi)
    float t = tanhf(c * (x + 0.044715f * x * x * x));
    return 0.5f * x * (1.0f + t);
}

// ---------------------------------------------------------------------------
// Elementwise f32 -> bf16 conversion
// ---------------------------------------------------------------------------
__global__ void k_f32_to_bf16(const float* __restrict__ in, bf16_t* __restrict__ out, long n) {
    long i = (long)blockIdx.x * blockDim.x + threadIdx.x;
    long stride = (long)gridDim.x * blockDim.x;
    for (; i < n; i += stride) out[i] = (bf16_t)in[i];
}

// ---------------------------------------------------------------------------
// Transpose W[K][N] (f32) -> Wt[N][K] (bf16), 32x32 LDS tiles
// ---------------------------------------------------------------------------
__global__ void k_transpose_bf16(const float* __restrict__ W, bf16_t* __restrict__ Wt,
                                 int K, int N) {
    __shared__ float tile[32][33];
    int n0 = blockIdx.x * 32, k0 = blockIdx.y * 32;
    int tx = threadIdx.x, ty = threadIdx.y;      // block (32, 8)
    for (int r = ty; r < 32; r += 8)
        tile[r][tx] = W[(size_t)(k0 + r) * N + n0 + tx];
    __syncthreads();
    for (int r = ty; r < 32; r += 8)
        Wt[(size_t)(n0 + r) * K + k0 + tx] = (bf16_t)tile[tx][r];
}

// ---------------------------------------------------------------------------
// ZOH discretization: Abar = exp(Lam*dt), Bbar = ((Abar-1)/Lam) * (B_re+iB_im)
// ---------------------------------------------------------------------------
__global__ void k_bbar(const float* __restrict__ Lam_re, const float* __restrict__ Lam_im,
                       const float* __restrict__ log_step,
                       const float* __restrict__ B_re, const float* __restrict__ B_im,
                       bf16_t* __restrict__ Bbar_re, bf16_t* __restrict__ Bbar_im,
                       float* __restrict__ abar_re, float* __restrict__ abar_im,
                       int P, int D) {
    long n = (long)P * D;
    long i = (long)blockIdx.x * blockDim.x + threadIdx.x;
    long stride = (long)gridDim.x * blockDim.x;
    for (; i < n; i += stride) {
        int p = (int)(i / D);
        int d = (int)(i % D);
        float lr = fminf(Lam_re[p], -1e-4f);
        float li = Lam_im[p];
        float dt = expf(log_step[p]);
        float er = expf(lr * dt);
        float ar = er * cosf(li * dt);
        float ai = er * sinf(li * dt);
        float den = lr * lr + li * li;
        float nr = ar - 1.0f, ni = ai;
        float cr = (nr * lr + ni * li) / den;
        float ci = (ni * lr - nr * li) / den;
        float br = B_re[i], bi = B_im[i];
        Bbar_re[i] = (bf16_t)(cr * br - ci * bi);
        Bbar_im[i] = (bf16_t)(cr * bi + ci * br);
        if (d == 0) { abar_re[p] = ar; abar_im[p] = ai; }
    }
}

// reset[b][l] = (s[b][l] != s[b][l-1]), with s[-1] := 0 (jnp.diff prepend=0)
__global__ void k_reset(const int* __restrict__ s, int* __restrict__ reset, int B, int L) {
    long n = (long)B * L;
    long i = (long)blockIdx.x * blockDim.x + threadIdx.x;
    if (i >= n) return;
    int l = (int)(i % L);
    int cur = s[i];
    int prev = (l == 0) ? 0 : s[i - 1];
    reset[i] = (cur != prev) ? 1 : 0;
}

// vx = v * x1 from packed p3 = [x1 | x2 | v]
__global__ void k_gate1(const float* __restrict__ p3, float* __restrict__ vxf,
                        bf16_t* __restrict__ vxb, long NBL, int Dm) {
    long n = NBL * Dm;
    long i = (long)blockIdx.x * blockDim.x + threadIdx.x;
    long stride = (long)gridDim.x * blockDim.x;
    for (; i < n; i += stride) {
        long row = i / Dm;
        int d = (int)(i % Dm);
        const float* pr = p3 + row * 3 * Dm;
        float val = pr[2 * Dm + d] * pr[d];
        vxf[i] = val;
        vxb[i] = (bf16_t)val;
    }
}

// g = gelu(y + D*vx) * x2
__global__ void k_gate2(const float* __restrict__ y, const float* __restrict__ vxf,
                        const float* __restrict__ p3, const float* __restrict__ Dvec,
                        bf16_t* __restrict__ g, long NBL, int Dm) {
    long n = NBL * Dm;
    long i = (long)blockIdx.x * blockDim.x + threadIdx.x;
    long stride = (long)gridDim.x * blockDim.x;
    for (; i < n; i += stride) {
        long row = i / Dm;
        int d = (int)(i % Dm);
        float x2 = p3[row * 3 * Dm + Dm + d];
        g[i] = (bf16_t)(gelu_f(y[i] + Dvec[d] * vxf[i]) * x2);
    }
}

// ---------------------------------------------------------------------------
// Reset-aware complex diagonal scan, in place on (bu_re, bu_im).
// ---------------------------------------------------------------------------
__global__ void k_scan(float* __restrict__ bu_re, float* __restrict__ bu_im,
                       const float* __restrict__ abar_re, const float* __restrict__ abar_im,
                       const int* __restrict__ reset, int Bn, int L, int P) {
    int t = blockIdx.x * blockDim.x + threadIdx.x;
    if (t >= Bn * P) return;
    int b = t / P, p = t % P;
    float ar = abar_re[p], ai = abar_im[p];
    const int* rs = reset + (size_t)b * L;
    size_t base = (size_t)b * L * P + p;
    float xr = 0.f, xi = 0.f;
    for (int l = 0; l < L; ++l) {
        size_t idx = base + (size_t)l * P;
        float br = bu_re[idx], bi = bu_im[idx];
        if (rs[l]) { xr = br; xi = bi; }
        else {
            float nr = ar * xr - ai * xi + br;
            float ni = ar * xi + ai * xr + bi;
            xr = nr; xi = ni;
        }
        bu_re[idx] = xr;
        bu_im[idx] = xi;
    }
}

// ---------------------------------------------------------------------------
// bf16 WMMA GEMM with async-to-LDS double-buffered tiles.
//   out[M][N] = A[M][K] @ Wt[N][K]^T (+bias) with fused (template) epilogues.
// Block tile 128x128 (8 waves, each 32x64 = 2x4 wmma accumulators).
// LDS tiles: 128 rows x 32 K-elems (64B data + 16B pad = 80B row stride),
// filled with GLOBAL_LOAD_ASYNC_TO_LDS_B128 (ASYNCcnt), double buffered.
// ---------------------------------------------------------------------------
enum { EPI_NONE = 0, EPI_SUBFROM = 1, EPI_RESID = 2, EPI_GELU = 3, EPI_FOURIER = 4 };

#define TILE_ROW_B 80           // bytes per LDS tile row (64 data + 16 pad)
#define TILE_BYTES (128 * TILE_ROW_B)

// Issue async copies of a 128x32 bf16 tile (8KB) into LDS: 2 x 16B per thread.
DEV void async_copy_tile(const bf16_t* __restrict__ src, int ldK, int rcBase, int k0,
                         unsigned ldsBase, int tid) {
#pragma unroll
    for (int c2 = 0; c2 < 2; ++c2) {
        int chunk = tid * 2 + c2;        // 0..511
        int row = chunk >> 2;            // 0..127
        int c = chunk & 3;               // 16B chunk within row
        unsigned long long ga =
            (unsigned long long)(src + (size_t)(rcBase + row) * ldK + k0 + c * 8);
        unsigned loff = ldsBase + row * TILE_ROW_B + c * 16;
        asm volatile("global_load_async_to_lds_b128 %0, %1, off"
                     :: "v"(loff), "v"(ga) : "memory");
    }
}

DEV v16bf fragA_lds(const char* tile, int r0, int lane) {
    int half = lane >> 4, m = lane & 15;
    const char* p = tile + (r0 + m) * TILE_ROW_B + half * 16;  // K {0..7}/{8..15}
    v8bf lo = *(const v8bf*)(p);
    v8bf hi = *(const v8bf*)(p + 32);                          // K {16..23}/{24..31}
    v16bf f;
#pragma unroll
    for (int i = 0; i < 8; ++i) { f[i] = lo[i]; f[i + 8] = hi[i]; }
    return f;
}

DEV v16bf fragB_lds(const char* tile, int n0, int lane) {
    int half = lane >> 4, n = lane & 15;
    const char* p = tile + (n0 + n) * TILE_ROW_B + half * 16;  // K {0..7}/{8..15}, then {16..23}/{24..31}
    v8bf lo = *(const v8bf*)(p);
    v8bf hi = *(const v8bf*)(p + 32);
    v16bf f;
#pragma unroll
    for (int i = 0; i < 8; ++i) { f[i] = lo[i]; f[i + 8] = hi[i]; }
    return f;
}

template <int EPI, bool BIAS, bool OF, bool OB>
__global__ __launch_bounds__(256) void k_gemm_bf16(
    const bf16_t* __restrict__ A, const bf16_t* __restrict__ Wt,
    const float* __restrict__ bias, const float* __restrict__ resid,
    float* __restrict__ outF, bf16_t* __restrict__ outB,
    int M, int N, int K) {
    __shared__ __align__(16) char smem[4 * TILE_BYTES];  // A0,A1,B0,B1 (40KB)

    const int tid = threadIdx.x;
    const int lane = tid & 31;
    const int wid = tid >> 5;
    const int wm = wid & 3, wn = wid >> 2;
    const int rowBlock = blockIdx.y * 128;
    const int colBlock = blockIdx.x * 128;
    const int rowBase = rowBlock + wm * 32;
    const int colBase = colBlock + wn * 64;
    const unsigned ldsBase = (unsigned)(uintptr_t)&smem[0];

    v8f zero = {0.f, 0.f, 0.f, 0.f, 0.f, 0.f, 0.f, 0.f};
    v8f acc[2][4];
#pragma unroll
    for (int i = 0; i < 2; ++i)
#pragma unroll
        for (int j = 0; j < 4; ++j) acc[i][j] = zero;

    // Prologue: async-fill buffer 0
    async_copy_tile(A, K, rowBlock, 0, ldsBase, tid);
    async_copy_tile(Wt, K, colBlock, 0, ldsBase + 2 * TILE_BYTES, tid);

    int cur = 0;
    for (int k0 = 0; k0 < K; k0 += 32) {
        asm volatile("s_wait_asynccnt 0x0" ::: "memory");
        __syncthreads();
        if (k0 + 32 < K) {  // async-fill the other buffer for the next K-step
            async_copy_tile(A, K, rowBlock, k0 + 32, ldsBase + (cur ^ 1) * TILE_BYTES, tid);
            async_copy_tile(Wt, K, colBlock, k0 + 32,
                            ldsBase + (2 + (cur ^ 1)) * TILE_BYTES, tid);
        }
        if (k0 + 64 < K) {  // warm L2 for the tile after next (global_prefetch_b8)
            __builtin_prefetch(A + (size_t)(rowBlock + (tid >> 1)) * K + k0 + 64);
            __builtin_prefetch(Wt + (size_t)(colBlock + (tid >> 1)) * K + k0 + 64);
        }

        const char* tA = smem + cur * TILE_BYTES;
        const char* tB = smem + (2 + cur) * TILE_BYTES;
        v16bf a0 = fragA_lds(tA, wm * 32, lane);
        v16bf a1 = fragA_lds(tA, wm * 32 + 16, lane);
        v16bf b0 = fragB_lds(tB, wn * 64 + 0, lane);
        v16bf b1 = fragB_lds(tB, wn * 64 + 16, lane);
        v16bf b2 = fragB_lds(tB, wn * 64 + 32, lane);
        v16bf b3 = fragB_lds(tB, wn * 64 + 48, lane);

        acc[0][0] = __builtin_amdgcn_wmma_f32_16x16x32_bf16(false, a0, false, b0, (short)0, acc[0][0], false, false);
        acc[0][1] = __builtin_amdgcn_wmma_f32_16x16x32_bf16(false, a0, false, b1, (short)0, acc[0][1], false, false);
        acc[0][2] = __builtin_amdgcn_wmma_f32_16x16x32_bf16(false, a0, false, b2, (short)0, acc[0][2], false, false);
        acc[0][3] = __builtin_amdgcn_wmma_f32_16x16x32_bf16(false, a0, false, b3, (short)0, acc[0][3], false, false);
        acc[1][0] = __builtin_amdgcn_wmma_f32_16x16x32_bf16(false, a1, false, b0, (short)0, acc[1][0], false, false);
        acc[1][1] = __builtin_amdgcn_wmma_f32_16x16x32_bf16(false, a1, false, b1, (short)0, acc[1][1], false, false);
        acc[1][2] = __builtin_amdgcn_wmma_f32_16x16x32_bf16(false, a1, false, b2, (short)0, acc[1][2], false, false);
        acc[1][3] = __builtin_amdgcn_wmma_f32_16x16x32_bf16(false, a1, false, b3, (short)0, acc[1][3], false, false);

        cur ^= 1;
    }

    // Epilogue (branch-free per template): VGPR v -> row (v + 8*half), lane&15 -> col
    const int half = lane >> 4, c16 = lane & 15;
#pragma unroll
    for (int i = 0; i < 2; ++i) {
#pragma unroll
        for (int j = 0; j < 4; ++j) {
            int col = colBase + j * 16 + c16;
            float bv = BIAS ? bias[col] : 0.f;
#pragma unroll
            for (int v = 0; v < 8; ++v) {
                int row = rowBase + i * 16 + half * 8 + v;
                size_t idx = (size_t)row * N + col;
                float val = acc[i][j][v] + bv;
                if constexpr (EPI == EPI_SUBFROM) {
                    outF[idx] = outF[idx] - val;
                } else {
                    if constexpr (EPI == EPI_RESID)   val += resid[idx];
                    if constexpr (EPI == EPI_GELU)    val = gelu_f(val);
                    if constexpr (EPI == EPI_FOURIER) val = (col & 1) ? sinf(val) : cosf(val);
                    if constexpr (OF) outF[idx] = val;
                    if constexpr (OB) outB[idx] = (bf16_t)val;
                }
            }
        }
    }
}

template <int EPI, bool BIAS, bool OF, bool OB>
static void launch_gemm(const bf16_t* A, const bf16_t* Wt, const float* bias,
                        const float* resid, float* oF, bf16_t* oB,
                        int M, int N, int K, hipStream_t st) {
    dim3 g(N / 128, M / 128);
    k_gemm_bf16<EPI, BIAS, OF, OB><<<g, dim3(256), 0, st>>>(A, Wt, bias, resid, oF, oB, M, N, K);
}

// ---------------------------------------------------------------------------
// Fused layernorm + fourier-modulation: out = standardize(zf)*scale + shift
// ---------------------------------------------------------------------------
__global__ __launch_bounds__(256) void k_final(const float* __restrict__ zf,
                                               const float* __restrict__ sh,
                                               const float* __restrict__ sc,
                                               float* __restrict__ out, int Dm) {
    __shared__ float s1[256], s2[256];
    int row = blockIdx.x;
    const float* zr = zf + (size_t)row * Dm;
    float a = 0.f, b = 0.f;
    for (int d = threadIdx.x; d < Dm; d += 256) {
        float v = zr[d];
        a += v; b += v * v;
    }
    s1[threadIdx.x] = a; s2[threadIdx.x] = b;
    __syncthreads();
    for (int off = 128; off > 0; off >>= 1) {
        if (threadIdx.x < off) {
            s1[threadIdx.x] += s1[threadIdx.x + off];
            s2[threadIdx.x] += s2[threadIdx.x + off];
        }
        __syncthreads();
    }
    float mu = s1[0] / Dm;
    float var = s2[0] / Dm - mu * mu;
    float rstd = rsqrtf(var + 1e-5f);
    for (int d = threadIdx.x; d < Dm; d += 256) {
        size_t i = (size_t)row * Dm + d;
        out[i] = (zr[d] - mu) * rstd * sc[i] + sh[i];
    }
}

// ---------------------------------------------------------------------------
// Host-side orchestration
// ---------------------------------------------------------------------------
extern "C" void kernel_launch(void* const* d_in, const int* in_sizes, int n_in,
                              void* d_out, int out_size, void* d_ws, size_t ws_size,
                              hipStream_t stream) {
    (void)in_sizes; (void)n_in; (void)out_size; (void)ws_size;
    const int Bn = 4, L = 2048, Dm = 1024, Pm = 1024;
    const long NBL = (long)Bn * L;                       // 8192
    const size_t F_PLANE  = (size_t)NBL * Dm * 4;        // 32 MB
    const size_t BF_PLANE = (size_t)NBL * Dm * 2;        // 16 MB

    const float* u        = (const float*)d_in[0];
    const float* r        = (const float*)d_in[1];
    const int*   s        = (const int*)  d_in[2];
    const float* shift_W  = (const float*)d_in[3];
    const float* shift_b  = (const float*)d_in[4];
    const float* scale_W  = (const float*)d_in[5];
    const float* scale_b  = (const float*)d_in[6];
    const float* in_W     = (const float*)d_in[7];
    const float* in_b     = (const float*)d_in[8];
    const float* out_W    = (const float*)d_in[9];
    const float* out_b    = (const float*)d_in[10];
    const float* Lam_re   = (const float*)d_in[11];
    const float* Lam_im   = (const float*)d_in[12];
    const float* B_re     = (const float*)d_in[13];
    const float* B_im     = (const float*)d_in[14];
    const float* C_re     = (const float*)d_in[15];
    const float* C_im     = (const float*)d_in[16];
    const float* Dvec     = (const float*)d_in[17];
    const float* log_step = (const float*)d_in[18];
    const float* W1       = (const float*)d_in[19];
    const float* b1       = (const float*)d_in[20];
    const float* W2       = (const float*)d_in[21];
    const float* b2       = (const float*)d_in[22];
    float* out = (float*)d_out;

    char* ws = (char*)d_ws;
    size_t off = 0;
    auto take = [&](size_t bytes) -> char* {
        char* p = ws + off;
        off += (bytes + 255) & ~(size_t)255;
        return p;
    };

    // Persistent (whole call)
    bf16_t* inWt   = (bf16_t*)take((size_t)3 * Dm * Dm * 2);
    bf16_t* outWt  = (bf16_t*)take((size_t)Dm * Dm * 2);
    bf16_t* W1t    = (bf16_t*)take((size_t)2 * Dm * Dm * 2);
    bf16_t* W2t    = (bf16_t*)take((size_t)2 * Dm * Dm * 2);
    bf16_t* shWt   = (bf16_t*)take((size_t)Dm * Dm * 2);
    bf16_t* scWt   = (bf16_t*)take((size_t)Dm * Dm * 2);
    bf16_t* Bre_bf = (bf16_t*)take((size_t)Pm * Dm * 2);
    bf16_t* Bim_bf = (bf16_t*)take((size_t)Pm * Dm * 2);
    bf16_t* Cre_bf = (bf16_t*)take((size_t)Dm * Pm * 2);
    bf16_t* Cim_bf = (bf16_t*)take((size_t)Dm * Pm * 2);
    float*  abar_re = (float*)take((size_t)Pm * 4);
    float*  abar_im = (float*)take((size_t)Pm * 4);
    int*    reset   = (int*)  take((size_t)Bn * L * 4);
    bf16_t* u_bf    = (bf16_t*)take(BF_PLANE);
    bf16_t* r_bf    = (bf16_t*)take(BF_PLANE);

    // Overlaid arena slots (lifetimes disjoint per-stage)
    char* A0 = take((size_t)NBL * 3 * Dm * 4);  // p3 (96MB)   -> later sh, sc
    char* A1 = take(F_PLANE);                   // vxf
    char* A2 = take(BF_PLANE);                  // vx_bf       -> g_bf
    char* A3 = take(F_PLANE);                   // bu/xs_re    -> z
    char* A4 = take(F_PLANE);                   // bu/xs_im    -> z_bf
    char* A5 = take(BF_PLANE);                  // xs_re_bf    -> h1_bf (spans A5+A6)
    char* A6 = take(BF_PLANE);                  // xs_im_bf
    char* A7 = take(F_PLANE);                   // y           -> zf

    float*  p3      = (float*)A0;
    float*  sh      = (float*)A0;
    float*  sc      = (float*)(A0 + F_PLANE);
    float*  vxf     = (float*)A1;
    bf16_t* vx_bf   = (bf16_t*)A2;
    bf16_t* g_bf    = (bf16_t*)A2;
    float*  bu_re   = (float*)A3;
    float*  z       = (float*)A3;
    float*  bu_im   = (float*)A4;
    bf16_t* z_bf    = (bf16_t*)A4;
    bf16_t* xsre_bf = (bf16_t*)A5;
    bf16_t* h1_bf   = (bf16_t*)A5;   // 32MB across A5+A6
    bf16_t* xsim_bf = (bf16_t*)A6;
    float*  y       = (float*)A7;
    float*  zf      = (float*)A7;

    dim3 tb(256);
    auto cblocks = [](long n) { return dim3((unsigned)((n + 255) / 256)); };
    dim3 tpb_t(32, 8);

    // ---- weight / parameter prep ----
    k_bbar<<<cblocks((long)Pm * Dm), tb, 0, stream>>>(Lam_re, Lam_im, log_step, B_re, B_im,
                                                      Bre_bf, Bim_bf, abar_re, abar_im, Pm, Dm);
    k_reset<<<cblocks((long)Bn * L), tb, 0, stream>>>(s, reset, Bn, L);
    k_f32_to_bf16<<<cblocks(NBL * Dm), tb, 0, stream>>>(u, u_bf, NBL * Dm);
    k_f32_to_bf16<<<cblocks(NBL * Dm), tb, 0, stream>>>(r, r_bf, NBL * Dm);
    k_f32_to_bf16<<<cblocks((long)Dm * Pm), tb, 0, stream>>>(C_re, Cre_bf, (long)Dm * Pm);
    k_f32_to_bf16<<<cblocks((long)Dm * Pm), tb, 0, stream>>>(C_im, Cim_bf, (long)Dm * Pm);
    k_transpose_bf16<<<dim3(3 * Dm / 32, Dm / 32), tpb_t, 0, stream>>>(in_W, inWt, Dm, 3 * Dm);
    k_transpose_bf16<<<dim3(Dm / 32, Dm / 32), tpb_t, 0, stream>>>(out_W, outWt, Dm, Dm);
    k_transpose_bf16<<<dim3(2 * Dm / 32, Dm / 32), tpb_t, 0, stream>>>(W1, W1t, Dm, 2 * Dm);
    k_transpose_bf16<<<dim3(Dm / 32, 2 * Dm / 32), tpb_t, 0, stream>>>(W2, W2t, 2 * Dm, Dm);
    k_transpose_bf16<<<dim3(Dm / 32, Dm / 32), tpb_t, 0, stream>>>(shift_W, shWt, Dm, Dm);
    k_transpose_bf16<<<dim3(Dm / 32, Dm / 32), tpb_t, 0, stream>>>(scale_W, scWt, Dm, Dm);

    // ---- S5 operator ----
    launch_gemm<EPI_NONE, true, true, false>(u_bf, inWt, in_b, nullptr, p3, nullptr,
                                             (int)NBL, 3 * Dm, Dm, stream);
    k_gate1<<<cblocks(NBL * Dm), tb, 0, stream>>>(p3, vxf, vx_bf, NBL, Dm);
    launch_gemm<EPI_NONE, false, true, false>(vx_bf, Bre_bf, nullptr, nullptr, bu_re, nullptr,
                                              (int)NBL, Pm, Dm, stream);
    launch_gemm<EPI_NONE, false, true, false>(vx_bf, Bim_bf, nullptr, nullptr, bu_im, nullptr,
                                              (int)NBL, Pm, Dm, stream);
    k_scan<<<cblocks((long)Bn * Pm), tb, 0, stream>>>(bu_re, bu_im, abar_re, abar_im, reset,
                                                      Bn, L, Pm);
    k_f32_to_bf16<<<cblocks(NBL * Pm), tb, 0, stream>>>(bu_re, xsre_bf, NBL * Pm);
    k_f32_to_bf16<<<cblocks(NBL * Pm), tb, 0, stream>>>(bu_im, xsim_bf, NBL * Pm);
    launch_gemm<EPI_NONE, false, true, false>(xsre_bf, Cre_bf, nullptr, nullptr, y, nullptr,
                                              (int)NBL, Dm, Pm, stream);
    launch_gemm<EPI_SUBFROM, false, true, false>(xsim_bf, Cim_bf, nullptr, nullptr, y, nullptr,
                                                 (int)NBL, Dm, Pm, stream);
    k_gate2<<<cblocks(NBL * Dm), tb, 0, stream>>>(y, vxf, p3, Dvec, g_bf, NBL, Dm);
    launch_gemm<EPI_RESID, true, true, true>(g_bf, outWt, out_b, u, z, z_bf,
                                             (int)NBL, Dm, Dm, stream);          // z = u + att

    // ---- FFN + residual ----
    launch_gemm<EPI_GELU, true, false, true>(z_bf, W1t, b1, nullptr, nullptr, h1_bf,
                                             (int)NBL, 2 * Dm, Dm, stream);
    launch_gemm<EPI_RESID, true, true, false>(h1_bf, W2t, b2, z, zf, nullptr,
                                              (int)NBL, Dm, 2 * Dm, stream);     // zf = z + h

    // ---- fourier-modulation branches ----
    launch_gemm<EPI_FOURIER, true, true, false>(r_bf, shWt, shift_b, nullptr, sh, nullptr,
                                                (int)NBL, Dm, Dm, stream);
    launch_gemm<EPI_FOURIER, true, true, false>(r_bf, scWt, scale_b, nullptr, sc, nullptr,
                                                (int)NBL, Dm, Dm, stream);

    // ---- layernorm + modulate ----
    k_final<<<dim3((unsigned)NBL), tb, 0, stream>>>(zf, sh, sc, out, Dm);
}
